// NystromAttention_36258113912855
// MI455X (gfx1250) — matrix operand order repaired and checked by reading, not compiled, not run on hardware
//
#include <hip/hip_runtime.h>

// ---------------------------------------------------------------------------
// CDNA5 (gfx1250, wave32) Nystrom attention.
//  - bf16 WMMA for every matmul touching T=2048 (QKV GEMM, scores, FA, y, proj)
//  - Tensor Data Mover (tensor_load_to_lds) double-buffered tile staging in the
//    big GEMMs, synchronized with s_wait_tensorcnt
//  - f32 VALU Newton-Schulz for the numerically sensitive 64x64 pinv
// ---------------------------------------------------------------------------

typedef __bf16 bf16;
typedef __attribute__((ext_vector_type(16))) __bf16 v16bf;
typedef __attribute__((ext_vector_type(8)))  __bf16 v8bf;
typedef __attribute__((ext_vector_type(8)))  float  v8f;
typedef __attribute__((ext_vector_type(4)))  unsigned int u32x4;
typedef __attribute__((ext_vector_type(8)))  int i32x8;
typedef __attribute__((ext_vector_type(4)))  int i32x4;

#define B_DIM 2
#define T_DIM 2048
#define E_DIM 512
#define H_DIM 8
#define HD    64
#define BH_DIM (B_DIM * H_DIM)

#if defined(__gfx1250__) && __has_builtin(__builtin_amdgcn_tensor_load_to_lds) && \
    __has_builtin(__builtin_amdgcn_s_wait_tensorcnt)
#define USE_TDM 1
#else
#define USE_TDM 0
#endif

__device__ __forceinline__ v8f vzero8() {
  v8f z;
#pragma unroll
  for (int i = 0; i < 8; ++i) z[i] = 0.0f;
  return z;
}

__device__ __forceinline__ v8f wmma_bf16(v16bf a, v16bf b, v8f c) {
  // (neg_a, A, neg_b, B, c_mod, C, reuse_a, reuse_b)
  return __builtin_amdgcn_wmma_f32_16x16x32_bf16(false, a, false, b, (short)0, c,
                                                 false, false);
}

// A fragment 16x32 bf16 (MxK) from row-major [M,K], leading dim `ld` (elems).
// ISA layout: lanes 0-15 -> rows 0-15, K {0..7} then {16..23};
//             lanes 16-31 -> same rows, K {8..15} then {24..31}.
__device__ __forceinline__ v16bf load_frag_a(const bf16* base, int ld) {
  const int lane = threadIdx.x & 31;
  const bf16* p = base + (lane & 15) * ld + ((lane >> 4) << 3);
  v8bf lo = *(const v8bf*)p;
  v8bf hi = *(const v8bf*)(p + 16);
  v16bf r;
#pragma unroll
  for (int i = 0; i < 8; ++i) { r[i] = lo[i]; r[i + 8] = hi[i]; }
  return r;
}

// B fragment 32x16 bf16 (KxN) sourced from row-major [N,K] storage (i.e. B^T):
// lanes 0-15 -> cols 0-15, K {0..7} then {16..23}; lanes 16-31 -> same cols, K {8..15} then {24..31}.
__device__ __forceinline__ v16bf load_frag_b(const bf16* base, int ld) {
  const int lane = threadIdx.x & 31;
  const bf16* p = base + (lane & 15) * ld + ((lane >> 4) << 3);
  v8bf lo = *(const v8bf*)p;
  v8bf hi = *(const v8bf*)(p + 16);
  v16bf r;
#pragma unroll
  for (int i = 0; i < 8; ++i) { r[i] = lo[i]; r[i + 8] = hi[i]; }
  return r;
}

#if USE_TDM
__device__ __forceinline__ unsigned lds_addr_of(const void* p) {
  // generic -> LDS address space; ptrtoint of an AS(3) pointer is the LDS byte offset
  return (unsigned)(unsigned long long)(__attribute__((address_space(3))) const char*)p;
}

// TENSOR_LOAD_TO_LDS of a 2D tile (tileW x tileH elems, 2-byte elements) from a
// row-major tensor with row stride strideElems. LDS receives rows packed
// (tileW elems per row). D# layout per CDNA5 ISA 8.3/8.4; groups 2/3 zero (2D).
__device__ __forceinline__ void tdm_load_2d(const void* gsrc, unsigned lds_off,
                                            unsigned tileW, unsigned tileH,
                                            unsigned tensorW, unsigned tensorH,
                                            unsigned strideElems) {
  unsigned long long ga = (unsigned long long)(uintptr_t)gsrc;
  u32x4 g0;
  g0[0] = 1u;                                      // count=1 (valid), user mode
  g0[1] = lds_off;                                 // lds_addr (bytes)
  g0[2] = (unsigned)(ga & 0xffffffffu);            // global_addr[31:0]
  g0[3] = (unsigned)((ga >> 32) & 0x1ffffffu) |    // global_addr[56:32]
          (2u << 30);                              // type=2 ("image")
  i32x8 g1;
  g1[0] = (int)(1u << 16);                         // data_size=1 -> 2 bytes
  g1[1] = (int)((tensorW & 0xffffu) << 16);        // tensor_dim0[15:0]
  g1[2] = (int)(((tensorW >> 16) & 0xffffu) | ((tensorH & 0xffffu) << 16));
  g1[3] = (int)(((tensorH >> 16) & 0xffffu) | ((tileW & 0xffffu) << 16));
  g1[4] = (int)(tileH & 0xffffu);                  // tile_dim1 (tile_dim2=0)
  g1[5] = (int)strideElems;                        // tensor_dim0_stride[31:0]
  g1[6] = 0;                                       // stride hi / dim1_stride lo
  g1[7] = 0;
  i32x4 z4;
  z4[0] = 0; z4[1] = 0; z4[2] = 0; z4[3] = 0;
  i32x8 z8;
#pragma unroll
  for (int i = 0; i < 8; ++i) z8[i] = 0;
  // 6-arg toolchain form: (g0, g1, g2, g3, g4, cpol)
  __builtin_amdgcn_tensor_load_to_lds(g0, g1, z4, z4, z8, 0);
}
#endif

// ---------------------------------------------------------------------------
// Elementwise conversion / transpose kernels
// ---------------------------------------------------------------------------
__global__ void f32_to_bf16_kernel(const float* __restrict__ src,
                                   bf16* __restrict__ dst, int n) {
  int i = blockIdx.x * blockDim.x + threadIdx.x;
  if (i < n) dst[i] = (bf16)src[i];
}

// src [K,N] row-major f32 -> dst [N,K] row-major bf16
__global__ void transpose_to_bf16_kernel(const float* __restrict__ src,
                                         bf16* __restrict__ dst, int K, int N) {
  int i = blockIdx.x * blockDim.x + threadIdx.x;
  if (i < K * N) {
    int k = i / N, n = i - k * N;
    dst[(size_t)n * K + k] = (bf16)src[i];
  }
}

// ---------------------------------------------------------------------------
// bf16 GEMM: C[M,N] = A[M,K] @ Bt[N,K]^T + bias[N]  (f32 out)
// 128x128 workgroup tile, 8 waves (wave = 16 rows x 128 cols).
// A/B 128x32 K-slices staged in LDS; TDM double-buffered when available.
// ---------------------------------------------------------------------------
__global__ __launch_bounds__(256) void gemm_bf16_kernel(
    const bf16* __restrict__ A, const bf16* __restrict__ Bt,
    const float* __restrict__ bias, float* __restrict__ C,
    int Mtot, int Ndim, int K) {
  __shared__ alignas(16) bf16 sA[2][128 * 32];
  __shared__ alignas(16) bf16 sB[2][128 * 32];
  const int wave = threadIdx.x >> 5;
  const int lane = threadIdx.x & 31;
  const int mBase = blockIdx.x * 128;
  const int n0 = blockIdx.y * 128;
  const bf16* Ablk = A + (size_t)mBase * K;
  const bf16* Bblk = Bt + (size_t)n0 * K;
  v8f acc[8];
#pragma unroll
  for (int i = 0; i < 8; ++i) acc[i] = vzero8();
  const int nsteps = K >> 5;

#if USE_TDM
  if (wave == 0) {  // one wave drives the DMA engine; TENSORcnt is per-wave
    tdm_load_2d(Ablk, lds_addr_of(&sA[0][0]), 32u, 128u, (unsigned)K,
                (unsigned)Mtot, (unsigned)K);
    tdm_load_2d(Bblk, lds_addr_of(&sB[0][0]), 32u, 128u, (unsigned)K,
                (unsigned)Ndim, (unsigned)K);
  }
  for (int s = 0; s < nsteps; ++s) {
    const int cur = s & 1;
    if (wave == 0) {
      if (s + 1 < nsteps) {  // prefetch next K-slice into the other buffer
        const int k0n = (s + 1) << 5;
        tdm_load_2d(Ablk + k0n, lds_addr_of(&sA[cur ^ 1][0]), 32u, 128u,
                    (unsigned)K, (unsigned)Mtot, (unsigned)K);
        tdm_load_2d(Bblk + k0n, lds_addr_of(&sB[cur ^ 1][0]), 32u, 128u,
                    (unsigned)K, (unsigned)Ndim, (unsigned)K);
        __builtin_amdgcn_s_wait_tensorcnt((short)2);  // current buffer landed
      } else {
        __builtin_amdgcn_s_wait_tensorcnt((short)0);
      }
    }
    __syncthreads();  // publish staged tiles to all waves
    v16bf a = load_frag_a(&sA[cur][wave * 16 * 32], 32);
#pragma unroll
    for (int nt = 0; nt < 8; ++nt) {
      v16bf b = load_frag_b(&sB[cur][nt * 16 * 32], 32);
      acc[nt] = wmma_bf16(a, b, acc[nt]);
    }
    __syncthreads();  // release buffer before TDM overwrites it next iter
  }
#else
  for (int s = 0; s < nsteps; ++s) {
    const int k0 = s << 5;
    __syncthreads();
    // cooperative staging: 512 x b128 per tile, 256 threads
    for (int i = threadIdx.x; i < 512; i += 256) {
      const int r = i >> 2, c = (i & 3) << 3;
      *(v8bf*)&sA[0][(size_t)r * 32 + c] =
          *(const v8bf*)(Ablk + (size_t)r * K + k0 + c);
      *(v8bf*)&sB[0][(size_t)r * 32 + c] =
          *(const v8bf*)(Bblk + (size_t)r * K + k0 + c);
    }
    __syncthreads();
    v16bf a = load_frag_a(&sA[0][wave * 16 * 32], 32);
#pragma unroll
    for (int nt = 0; nt < 8; ++nt) {
      v16bf b = load_frag_b(&sB[0][nt * 16 * 32], 32);
      acc[nt] = wmma_bf16(a, b, acc[nt]);
    }
  }
#endif

  const int cn = lane & 15, half = lane >> 4;
  const int m0 = mBase + wave * 16;
#pragma unroll
  for (int nt = 0; nt < 8; ++nt) {
    const int n = n0 + nt * 16 + cn;
    const float bv = bias[n];
#pragma unroll
    for (int v = 0; v < 8; ++v) {
      const int m = m0 + v + half * 8;
      C[(size_t)m * Ndim + n] = acc[nt][v] + bv;
    }
  }
}

// ---------------------------------------------------------------------------
// Split heads: qkv f32 [B*T, 1536] -> q_bf (scaled 1/8), k_bf as [B,H,T,64] bf16
// ---------------------------------------------------------------------------
__global__ void split_heads_kernel(const float* __restrict__ qkv,
                                   bf16* __restrict__ q_bf,
                                   bf16* __restrict__ k_bf) {
  int i = blockIdx.x * blockDim.x + threadIdx.x;  // flat [B,H,T,HD]
  if (i >= B_DIM * H_DIM * T_DIM * HD) return;
  int d = i & 63;
  int t = (i >> 6) & (T_DIM - 1);
  int h = (i >> 17) & 7;
  int b = i >> 20;
  const float* row = qkv + ((size_t)(b * T_DIM + t)) * (3 * E_DIM);
  q_bf[i] = (bf16)(0.125f * row[h * HD + d]);  // 1/sqrt(64)
  k_bf[i] = (bf16)(row[E_DIM + h * HD + d]);
}

// ---------------------------------------------------------------------------
// Landmark means (group size 32); q landmark carries the 1/8 scale.
// ---------------------------------------------------------------------------
__global__ void landmarks_kernel(const float* __restrict__ qkv,
                                 float* __restrict__ ql_f, float* __restrict__ kl_f,
                                 bf16* __restrict__ ql_bf, bf16* __restrict__ kl_bf) {
  int i = blockIdx.x * blockDim.x + threadIdx.x;  // flat [B,H,64,64]
  if (i >= BH_DIM * 64 * 64) return;
  int d = i & 63;
  int m = (i >> 6) & 63;
  int h = (i >> 12) & 7;
  int b = i >> 15;
  const float* p = qkv + ((size_t)(b * T_DIM + m * 32)) * (3 * E_DIM) + h * HD + d;
  float sq = 0.f, sk = 0.f;
#pragma unroll 4
  for (int j = 0; j < 32; ++j) {
    sq += p[(size_t)j * (3 * E_DIM)];
    sk += p[(size_t)j * (3 * E_DIM) + E_DIM];
  }
  float qv = 0.125f * sq * (1.0f / 32.0f);
  float kv = sk * (1.0f / 32.0f);
  ql_f[i] = qv; kl_f[i] = kv;
  ql_bf[i] = (bf16)qv; kl_bf[i] = (bf16)kv;
}

// ---------------------------------------------------------------------------
// A = softmax(tril(q_l @ k_l^T)); pinvA via 6 Newton-Schulz iters (f32, LDS).
// Writes pinvA^T bf16. One workgroup (256 thr) per (b,h); ~50KB LDS.
// ---------------------------------------------------------------------------
__global__ __launch_bounds__(256) void pinv_kernel(const float* __restrict__ ql_f,
                                                   const float* __restrict__ kl_f,
                                                   bf16* __restrict__ pAt) {
  __shared__ float sA[64][65];
  __shared__ float sZ[64][65];
  __shared__ float sP[64][65];
  __shared__ float sScal;
  const int bh = blockIdx.x;
  const int tid = threadIdx.x;
  const float* ql = ql_f + (size_t)bh * 4096;
  const float* kl = kl_f + (size_t)bh * 4096;
  for (int i = tid; i < 4096; i += 256) {  // stage ql -> sZ, kl -> sP
    sZ[i >> 6][i & 63] = ql[i];
    sP[i >> 6][i & 63] = kl[i];
  }
  __syncthreads();
  const int ri = tid >> 2;
  const int j0 = (tid & 3) << 4;
  float tr[16];
  // S = ql @ kl^T
#pragma unroll
  for (int jj = 0; jj < 16; ++jj) {
    float s = 0.f;
    for (int d = 0; d < 64; ++d) s += sZ[ri][d] * sP[j0 + jj][d];
    tr[jj] = s;
  }
  __syncthreads();
#pragma unroll
  for (int jj = 0; jj < 16; ++jj) sA[ri][j0 + jj] = tr[jj];
  __syncthreads();
  // masked lower-triangular softmax
  if (tid < 64) {
    const int i = tid;
    float mx = -3.4e38f;
    for (int j = 0; j <= i; ++j) mx = fmaxf(mx, sA[i][j]);
    float sum = 0.f;
    for (int j = 0; j <= i; ++j) { float e = __expf(sA[i][j] - mx); sA[i][j] = e; sum += e; }
    float inv = 1.f / sum;
    for (int j = 0; j <= i; ++j) sA[i][j] *= inv;
    for (int j = i + 1; j < 64; ++j) sA[i][j] = 0.f;
  }
  __syncthreads();
  if (tid < 64) {
    float rs = 0.f, cs = 0.f;
    for (int j = 0; j < 64; ++j) { rs += fabsf(sA[tid][j]); cs += fabsf(sA[j][tid]); }
    sP[0][tid] = rs; sP[1][tid] = cs;
  }
  __syncthreads();
  if (tid == 0) {
    float mr = 0.f, mc = 0.f;
    for (int j = 0; j < 64; ++j) { mr = fmaxf(mr, sP[0][j]); mc = fmaxf(mc, sP[1][j]); }
    sScal = 1.f / (mr * mc);
  }
  __syncthreads();
  for (int i = tid; i < 4096; i += 256) {  // Z = A^T * scal
    int r = i >> 6, c = i & 63;
    sZ[r][c] = sA[c][r] * sScal;
  }
  __syncthreads();
  float xz[64];
  for (int it = 0; it < 6; ++it) {
    // xz row = (A @ Z)[ri][:]
#pragma unroll
    for (int d = 0; d < 64; ++d) xz[d] = 0.f;
    for (int k = 0; k < 64; ++k) {
      float a = sA[ri][k];
#pragma unroll
      for (int d = 0; d < 64; ++d) xz[d] += a * sZ[k][d];
    }
    __syncthreads();
#pragma unroll
    for (int jj = 0; jj < 16; ++jj) {  // P = 7I - XZ
      int j = j0 + jj;
      sP[ri][j] = ((ri == j) ? 7.0f : 0.0f) - xz[j];
    }
    __syncthreads();
#pragma unroll
    for (int jj = 0; jj < 16; ++jj) {  // T = XZ @ P
      float s = 0.f;
      for (int k = 0; k < 64; ++k) s += xz[k] * sP[k][j0 + jj];
      tr[jj] = s;
    }
    __syncthreads();
#pragma unroll
    for (int jj = 0; jj < 16; ++jj) {  // P = 15I - T
      int j = j0 + jj;
      sP[ri][j] = ((ri == j) ? 15.0f : 0.0f) - tr[jj];
    }
    __syncthreads();
#pragma unroll
    for (int jj = 0; jj < 16; ++jj) {  // T = XZ @ P
      float s = 0.f;
      for (int k = 0; k < 64; ++k) s += xz[k] * sP[k][j0 + jj];
      tr[jj] = s;
    }
    __syncthreads();
#pragma unroll
    for (int jj = 0; jj < 16; ++jj) {  // P = 13I - T
      int j = j0 + jj;
      sP[ri][j] = ((ri == j) ? 13.0f : 0.0f) - tr[jj];
    }
    __syncthreads();
#pragma unroll
    for (int jj = 0; jj < 16; ++jj) {  // T = Z @ P
      float s = 0.f;
      for (int k = 0; k < 64; ++k) s += sZ[ri][k] * sP[k][j0 + jj];
      tr[jj] = s;
    }
    __syncthreads();
#pragma unroll
    for (int jj = 0; jj < 16; ++jj) sZ[ri][j0 + jj] = 0.25f * tr[jj];  // Z = T/4
    __syncthreads();
  }
  for (int i = tid; i < 4096; i += 256) {  // pinvA^T: out[n*64+k] = Z[k][n]
    int n = i >> 6, k = i & 63;
    pAt[(size_t)bh * 4096 + i] = (bf16)sZ[k][n];
  }
}

// ---------------------------------------------------------------------------
// BV = softmax(tril(q_l @ k^T)) @ v.  The MxT tril mask kills all t>=64, so
// this is two 64x64x64 WMMA matmuls around a masked softmax.
// Writes BV^T bf16 ([d][m]). One workgroup per (b,h).
// ---------------------------------------------------------------------------
__global__ __launch_bounds__(256) void bv_kernel(const float* __restrict__ qkv,
                                                 const bf16* __restrict__ k_bf,
                                                 const bf16* __restrict__ ql_bf,
                                                 bf16* __restrict__ BVt) {
  __shared__ alignas(16) bf16 sVT[64][72];  // v[0:64]^T : [d][t]
  __shared__ float sS[64][65];
  __shared__ alignas(16) bf16 sBt[64][72];
  const int bh = blockIdx.x;
  const int b = bh >> 3, h = bh & 7;
  for (int i = threadIdx.x; i < 4096; i += 256) {
    int t = i >> 6, d = i & 63;
    sVT[d][t] = (bf16)qkv[((size_t)(b * T_DIM + t)) * (3 * E_DIM) + 2 * E_DIM + h * HD + d];
  }
  __syncthreads();
  const int wave = threadIdx.x >> 5;
  const int lane = threadIdx.x & 31;
  const int cn = lane & 15, half = lane >> 4;
  if (wave < 4) {  // wave-uniform branch: EXEC all ones inside
    const bf16* qlb = ql_bf + (size_t)bh * 4096 + wave * 16 * 64;
    const bf16* kb = k_bf + (size_t)bh * T_DIM * HD;
    v8f acc[4];
#pragma unroll
    for (int i = 0; i < 4; ++i) acc[i] = vzero8();
#pragma unroll
    for (int k0 = 0; k0 < 64; k0 += 32) {
      v16bf a = load_frag_a(qlb + k0, 64);
#pragma unroll
      for (int nt = 0; nt < 4; ++nt)
        acc[nt] = wmma_bf16(a, load_frag_b(kb + (size_t)nt * 16 * 64 + k0, 64), acc[nt]);
    }
#pragma unroll
    for (int nt = 0; nt < 4; ++nt)
#pragma unroll
      for (int v = 0; v < 8; ++v)
        sS[wave * 16 + v + half * 8][nt * 16 + cn] = acc[nt][v];
  }
  __syncthreads();
  if (threadIdx.x < 64) {
    const int m = threadIdx.x;
    const int lim = m + 1;  // valid t <= m
    float mx = -3.4e38f;
    for (int t = 0; t < lim; ++t) mx = fmaxf(mx, sS[m][t]);
    float sum = 0.f;
    for (int t = 0; t < lim; ++t) sum += __expf(sS[m][t] - mx);
    float inv = 1.f / sum;
    for (int t = 0; t < lim; ++t) sBt[m][t] = (bf16)(__expf(sS[m][t] - mx) * inv);
    for (int t = lim; t < 64; ++t) sBt[m][t] = (bf16)0.f;
  }
  __syncthreads();
  if (wave < 4) {
    v8f acc[4];
#pragma unroll
    for (int i = 0; i < 4; ++i) acc[i] = vzero8();
#pragma unroll
    for (int k0 = 0; k0 < 64; k0 += 32) {
      v16bf a = load_frag_a(&sBt[wave * 16][0] + k0, 72);
#pragma unroll
      for (int nt = 0; nt < 4; ++nt)
        acc[nt] = wmma_bf16(a, load_frag_b(&sVT[nt * 16][0] + k0, 72), acc[nt]);
    }
#pragma unroll
    for (int nt = 0; nt < 4; ++nt)
#pragma unroll
      for (int v = 0; v < 8; ++v) {
        int m = wave * 16 + v + half * 8;
        int d = nt * 16 + cn;
        BVt[(size_t)bh * 4096 + (size_t)d * 64 + m] = (bf16)acc[nt][v];
      }
  }
}

// ---------------------------------------------------------------------------
// F = softmax(tril(q @ k_l^T)); y = (F @ pinvA) @ BV.  Grid (T/128, BH).
// LDS: sF plus a pool shared by f32 scores (stages 1-2) and bf16 FA (3-4).
// ---------------------------------------------------------------------------
__global__ __launch_bounds__(256) void attn_kernel(const bf16* __restrict__ q_bf,
                                                   const bf16* __restrict__ kl_bf,
                                                   const bf16* __restrict__ pAt,
                                                   const bf16* __restrict__ BVt,
                                                   bf16* __restrict__ y_bf) {
  __shared__ alignas(16) bf16 sF[128][72];
  __shared__ alignas(16) char pool[128 * 65 * sizeof(float)];
  float(*sS)[65] = (float(*)[65])pool;
  bf16(*sFA)[72] = (bf16(*)[72])pool;

  const int bh = blockIdx.y;
  const int tile = blockIdx.x;
  const int wave = threadIdx.x >> 5;
  const int lane = threadIdx.x & 31;
  const int cn = lane & 15, half = lane >> 4;
  const int row0 = tile * 128 + wave * 16;

  {  // stage 1: scores = q @ k_l^T -> sS (f32)
    const bf16* qb = q_bf + ((size_t)bh * T_DIM + row0) * HD;
    const bf16* kl = kl_bf + (size_t)bh * 4096;
    v8f acc[4];
#pragma unroll
    for (int i = 0; i < 4; ++i) acc[i] = vzero8();
#pragma unroll
    for (int k0 = 0; k0 < 64; k0 += 32) {
      v16bf a = load_frag_a(qb + k0, HD);
#pragma unroll
      for (int nt = 0; nt < 4; ++nt)
        acc[nt] = wmma_bf16(a, load_frag_b(kl + (size_t)nt * 16 * 64 + k0, 64), acc[nt]);
    }
#pragma unroll
    for (int nt = 0; nt < 4; ++nt)
#pragma unroll
      for (int v = 0; v < 8; ++v)
        sS[wave * 16 + v + half * 8][nt * 16 + cn] = acc[nt][v];
  }
  __syncthreads();
  // stage 2: masked softmax over landmarks (keep m <= t) -> sF (bf16)
  if (threadIdx.x < 128) {
    const int r = threadIdx.x;
    const int t = tile * 128 + r;
    const int lim = (t < 63) ? (t + 1) : 64;
    float mx = -3.4e38f;
    for (int m = 0; m < lim; ++m) mx = fmaxf(mx, sS[r][m]);
    float sum = 0.f;
    for (int m = 0; m < lim; ++m) sum += __expf(sS[r][m] - mx);
    const float inv = 1.f / sum;
    for (int m = 0; m < lim; ++m) sF[r][m] = (bf16)(__expf(sS[r][m] - mx) * inv);
    for (int m = lim; m < 64; ++m) sF[r][m] = (bf16)0.f;
  }
  __syncthreads();
  {  // stage 3: FA = F @ pinvA -> sFA (reuses score pool)
    const bf16* pA = pAt + (size_t)bh * 4096;
    v8f acc[4];
#pragma unroll
    for (int i = 0; i < 4; ++i) acc[i] = vzero8();
#pragma unroll
    for (int k0 = 0; k0 < 64; k0 += 32) {
      v16bf a = load_frag_a(&sF[wave * 16][0] + k0, 72);
#pragma unroll
      for (int nt = 0; nt < 4; ++nt)
        acc[nt] = wmma_bf16(a, load_frag_b(pA + (size_t)nt * 16 * 64 + k0, 64), acc[nt]);
    }
#pragma unroll
    for (int nt = 0; nt < 4; ++nt)
#pragma unroll
      for (int v = 0; v < 8; ++v)
        sFA[wave * 16 + v + half * 8][nt * 16 + cn] = (bf16)acc[nt][v];
  }
  __syncthreads();
  {  // stage 4: y = FA @ BV -> y_bf [B*T, 512]
    const bf16* bv = BVt + (size_t)bh * 4096;
    v8f acc[4];
#pragma unroll
    for (int i = 0; i < 4; ++i) acc[i] = vzero8();
#pragma unroll
    for (int k0 = 0; k0 < 64; k0 += 32) {
      v16bf a = load_frag_a(&sFA[wave * 16][0] + k0, 72);
#pragma unroll
      for (int nt = 0; nt < 4; ++nt)
        acc[nt] = wmma_bf16(a, load_frag_b(bv + (size_t)nt * 16 * 64 + k0, 64), acc[nt]);
    }
    const int b = bh >> 3, h = bh & 7;
#pragma unroll
    for (int nt = 0; nt < 4; ++nt)
#pragma unroll
      for (int v = 0; v < 8; ++v) {
        const int t = row0 + v + half * 8;
        const int d = nt * 16 + cn;
        y_bf[((size_t)(b * T_DIM + t)) * E_DIM + h * HD + d] = (bf16)acc[nt][v];
      }
  }
}

// ---------------------------------------------------------------------------
// Host launch
// ---------------------------------------------------------------------------
extern "C" void kernel_launch(void* const* d_in, const int* in_sizes, int n_in,
                              void* d_out, int out_size, void* d_ws, size_t ws_size,
                              hipStream_t stream) {
  (void)in_sizes; (void)n_in; (void)out_size; (void)ws_size;
  const float* x      = (const float*)d_in[0];
  const float* w_qkv  = (const float*)d_in[1];
  const float* b_qkv  = (const float*)d_in[2];
  const float* w_proj = (const float*)d_in[3];
  const float* b_proj = (const float*)d_in[4];
  float* out = (float*)d_out;

  char* ws = (char*)d_ws;
  size_t off = 0;
  auto alloc = [&](size_t bytes) -> char* {
    char* p = ws + off;
    off += (bytes + 255) & ~(size_t)255;
    return p;
  };
  const size_t NT = (size_t)B_DIM * T_DIM;  // 4096
  bf16* x_bf   = (bf16*)alloc(NT * E_DIM * 2);
  bf16* wqkvT  = (bf16*)alloc((size_t)3 * E_DIM * E_DIM * 2);
  bf16* wprojT = (bf16*)alloc((size_t)E_DIM * E_DIM * 2);
  float* qkv   = (float*)alloc(NT * 3 * E_DIM * 4);
  bf16* q_bf   = (bf16*)alloc((size_t)BH_DIM * T_DIM * HD * 2);
  bf16* k_bf   = (bf16*)alloc((size_t)BH_DIM * T_DIM * HD * 2);
  float* ql_f  = (float*)alloc((size_t)BH_DIM * 4096 * 4);
  float* kl_f  = (float*)alloc((size_t)BH_DIM * 4096 * 4);
  bf16* ql_bf  = (bf16*)alloc((size_t)BH_DIM * 4096 * 2);
  bf16* kl_bf  = (bf16*)alloc((size_t)BH_DIM * 4096 * 2);
  bf16* pAt    = (bf16*)alloc((size_t)BH_DIM * 4096 * 2);
  bf16* BVt    = (bf16*)alloc((size_t)BH_DIM * 4096 * 2);
  bf16* y_bf   = (bf16*)alloc(NT * E_DIM * 2);

  {  // conversions / weight transposes
    int n = (int)(NT * E_DIM);
    f32_to_bf16_kernel<<<n / 256, 256, 0, stream>>>(x, x_bf, n);
    transpose_to_bf16_kernel<<<(E_DIM * 3 * E_DIM) / 256, 256, 0, stream>>>(
        w_qkv, wqkvT, E_DIM, 3 * E_DIM);
    transpose_to_bf16_kernel<<<(E_DIM * E_DIM) / 256, 256, 0, stream>>>(
        w_proj, wprojT, E_DIM, E_DIM);
  }
  // qkv = x @ w_qkv + b_qkv
  gemm_bf16_kernel<<<dim3((unsigned)(NT / 128), (3 * E_DIM) / 128), 256, 0, stream>>>(
      x_bf, wqkvT, b_qkv, qkv, (int)NT, 3 * E_DIM, E_DIM);
  {  // split heads (+scale q), landmarks
    int n = BH_DIM * T_DIM * HD;
    split_heads_kernel<<<n / 256, 256, 0, stream>>>(qkv, q_bf, k_bf);
    landmarks_kernel<<<(BH_DIM * 64 * 64) / 256, 256, 0, stream>>>(
        qkv, ql_f, kl_f, ql_bf, kl_bf);
  }
  pinv_kernel<<<BH_DIM, 256, 0, stream>>>(ql_f, kl_f, pAt);
  bv_kernel<<<BH_DIM, 256, 0, stream>>>(qkv, k_bf, ql_bf, BVt);
  attn_kernel<<<dim3(T_DIM / 128, BH_DIM), 256, 0, stream>>>(
      q_bf, kl_bf, pAt, BVt, y_bf);
  // out = y @ w_proj + b_proj
  gemm_bf16_kernel<<<dim3((unsigned)(NT / 128), E_DIM / 128), 256, 0, stream>>>(
      y_bf, wprojT, b_proj, out, (int)NT, E_DIM, E_DIM);
}
